// BMPHead_24670292148309
// MI455X (gfx1250) — matrix operand order repaired and checked by reading, not compiled, not run on hardware
//
#include <hip/hip_runtime.h>

typedef __bf16 v16bf __attribute__((ext_vector_type(16)));
typedef __bf16 v8bf  __attribute__((ext_vector_type(8)));
typedef float  v8f   __attribute__((ext_vector_type(8)));

#define BATCH 4

// ---------------------------------------------------------------------------
// Weight packing: OIHW fp32 -> bf16 B-fragments in wave-native layout.
// Fragment tile (kt, nt): 32 lanes x 16 bf16 contiguous.
//   lane l: N = nt*16 + (l&15), K = kt*32 + (l>>4)*16 + i  (i = 0..15)
// K index = tap*CinPad + ci, tap = dy*3+dx.
// ---------------------------------------------------------------------------
__global__ void pack_w3x3(const float* __restrict__ src, int Cin, int CinPad,
                          int Ktiles, int Ntiles, __bf16* __restrict__ dst) {
    int e = blockIdx.x * blockDim.x + threadIdx.x;
    int total = Ktiles * Ntiles * 512;
    if (e >= total) return;
    int i    = e & 15;
    int lane = (e >> 4) & 31;
    int tile = e >> 9;
    int nt = tile % Ntiles, kt = tile / Ntiles;
    int n = nt * 16 + (lane & 15);
    int k = kt * 32 + (lane >> 4) * 16 + i;
    int tap = k / CinPad, ci = k % CinPad;
    float v = 0.0f;
    if (ci < Cin)
        v = src[(((size_t)n * Cin + ci) * 3 + tap / 3) * 3 + tap % 3];
    dst[e] = (__bf16)v;
}

// 1x1 head weights: src (157,256), packed K=256 (8 tiles) x N=160 (10 tiles)
__global__ void pack_head(const float* __restrict__ src, __bf16* __restrict__ dst) {
    int e = blockIdx.x * blockDim.x + threadIdx.x;
    int total = 8 * 10 * 512;
    if (e >= total) return;
    int i    = e & 15;
    int lane = (e >> 4) & 31;
    int tile = e >> 9;
    int nt = tile % 10, kt = tile / 10;
    int n = nt * 16 + (lane & 15);
    int k = kt * 32 + (lane >> 4) * 16 + i;
    float v = (n < 157) ? src[(size_t)n * 256 + k] : 0.0f;
    dst[e] = (__bf16)v;
}

// Zero-fill the padding page used for branchless border taps.
__global__ void zero_fill_bf16(__bf16* __restrict__ p, int n) {
    int i = blockIdx.x * blockDim.x + threadIdx.x;
    if (i < n) p[i] = (__bf16)0.0f;
}

// ---------------------------------------------------------------------------
// Align-corners bilinear resize, NCHW fp32 -> fp32 (f0: 200->100, f4: 13->25)
// ---------------------------------------------------------------------------
__global__ void resize_ac_nchw(const float* __restrict__ src, int H, int W,
                               int OH, int OW, int BC, float* __restrict__ dst) {
    int idx = blockIdx.x * blockDim.x + threadIdx.x;
    int total = BC * OH * OW;
    if (idx >= total) return;
    int ox = idx % OW;
    int oy = (idx / OW) % OH;
    int bc = idx / (OW * OH);
    float sy = (OH > 1) ? oy * (float)(H - 1) / (float)(OH - 1) : 0.0f;
    float sx = (OW > 1) ? ox * (float)(W - 1) / (float)(OW - 1) : 0.0f;
    int y0 = (int)floorf(sy), x0 = (int)floorf(sx);
    int y1 = min(y0 + 1, H - 1), x1 = min(x0 + 1, W - 1);
    float wy = sy - (float)y0, wx = sx - (float)x0;
    const float* s = src + (size_t)bc * H * W;
    float v = s[y0 * W + x0] * (1.f - wy) * (1.f - wx)
            + s[y0 * W + x1] * (1.f - wy) * wx
            + s[y1 * W + x0] * wy * (1.f - wx)
            + s[y1 * W + x1] * wy * wx;
    dst[idx] = v;
}

// ---------------------------------------------------------------------------
// Resize level input to g x g, channel-last bf16, C padded to 288:
//   c<256: bilinear(x); c==256: xx ramp; c==257: yy ramp; c>=258: 0
// (align-corners bilinear of a linear ramp is exact -> coords = linspace(-1,1,g))
// ---------------------------------------------------------------------------
__global__ void resize_grid(const float* __restrict__ src, int h, int w, int g,
                            __bf16* __restrict__ dst) {
    int idx = blockIdx.x * blockDim.x + threadIdx.x;
    int total = BATCH * g * g * 288;
    if (idx >= total) return;
    int c   = idx % 288;
    int pix = idx / 288;
    int gx = pix % g;
    int gy = (pix / g) % g;
    int b  = pix / (g * g);
    float v;
    if (c < 256) {
        float sy = gy * (float)(h - 1) / (float)(g - 1);
        float sx = gx * (float)(w - 1) / (float)(g - 1);
        int y0 = (int)floorf(sy), x0 = (int)floorf(sx);
        int y1 = min(y0 + 1, h - 1), x1 = min(x0 + 1, w - 1);
        float wy = sy - (float)y0, wx = sx - (float)x0;
        const float* s = src + ((size_t)b * 256 + c) * h * w;
        v = s[y0 * w + x0] * (1.f - wy) * (1.f - wx)
          + s[y0 * w + x1] * (1.f - wy) * wx
          + s[y1 * w + x0] * wy * (1.f - wx)
          + s[y1 * w + x1] * wy * wx;
    } else if (c == 256) {
        v = -1.0f + 2.0f * (float)gx / (float)(g - 1);
    } else if (c == 257) {
        v = -1.0f + 2.0f * (float)gy / (float)(g - 1);
    } else {
        v = 0.0f;
    }
    dst[idx] = (__bf16)v;
}

// ---------------------------------------------------------------------------
// 3x3 pad-1 conv as implicit GEMM via v_wmma_f32_16x16x32_bf16.
// Register blocking: one wave computes 32(M) x 32(N) = 4 accumulators; each A
// fragment feeds 2 WMMAs and each B fragment feeds 2 WMMAs. Border taps use a
// zero page instead of predication so all fragment loads are unconditional and
// the fully-unrolled CHUNKS loop lets the compiler batch loads ahead of WMMAs.
// in : (B, g, g, cstride) bf16 channel-last;  out: (B, g, g, out_cstride) bf16
// wf : packed B fragments, K = 9*CHUNKS*32, N = 256 (NT=16)
// ---------------------------------------------------------------------------
template <int CHUNKS>
__global__ void __launch_bounds__(256)
conv3x3_wmma(const __bf16* __restrict__ in, int g, int cstride,
             const __bf16* __restrict__ wf, const __bf16* __restrict__ zpage,
             const float* __restrict__ bias,
             __bf16* __restrict__ out, int out_cstride, int relu) {
    const int NT = 16;
    int wid  = (blockIdx.x * blockDim.x + threadIdx.x) >> 5;
    int lane = threadIdx.x & 31;
    int mtp = wid >> 3;          // pair of M tiles
    int ntp = wid & 7;           // pair of N tiles
    int row = lane & 15, khalf = lane >> 4;

    // two pixel rows handled by this lane
    int p0 = mtp * 32 + row;
    int p1 = p0 + 16;
    int gg = g * g;
    int b0 = p0 / gg, r0 = p0 - b0 * gg, y0 = r0 / g, x0 = r0 - y0 * g;
    int b1 = p1 / gg, r1 = p1 - b1 * gg, y1 = r1 / g, x1 = r1 - y1 * g;

    v8f acc00 = {}, acc01 = {}, acc10 = {}, acc11 = {};
    const int koff = khalf * 16;

    for (int tap = 0; tap < 9; ++tap) {
        int dy = tap / 3 - 1, dx = tap % 3 - 1;
        int ny0 = y0 + dy, nx0 = x0 + dx;
        int ny1 = y1 + dy, nx1 = x1 + dx;
        bool v0 = ((unsigned)ny0 < (unsigned)g) && ((unsigned)nx0 < (unsigned)g);
        bool v1 = ((unsigned)ny1 < (unsigned)g) && ((unsigned)nx1 < (unsigned)g);
        const __bf16* rp0 = v0 ? in + (size_t)((b0 * g + ny0) * g + nx0) * cstride : zpage;
        const __bf16* rp1 = v1 ? in + (size_t)((b1 * g + ny1) * g + nx1) * cstride : zpage;
        const __bf16* wp  = wf + (((size_t)(tap * CHUNKS) * NT + ntp * 2) * 32 + lane) * 16;
        if (tap < 8)   // stream next tap's weight tiles
            __builtin_prefetch(wf + (((size_t)((tap + 1) * CHUNKS) * NT + ntp * 2) * 32 + lane) * 16, 0, 0);
#pragma unroll
        for (int c = 0; c < CHUNKS; ++c) {
            v8bf a0l = *(const v8bf*)(rp0 + c * 32 + koff);
            v8bf a0h = *(const v8bf*)(rp0 + c * 32 + koff + 8);
            v8bf a1l = *(const v8bf*)(rp1 + c * 32 + koff);
            v8bf a1h = *(const v8bf*)(rp1 + c * 32 + koff + 8);
            v16bf a0 = __builtin_shufflevector(a0l, a0h,
                0, 1, 2, 3, 4, 5, 6, 7, 8, 9, 10, 11, 12, 13, 14, 15);
            v16bf a1 = __builtin_shufflevector(a1l, a1h,
                0, 1, 2, 3, 4, 5, 6, 7, 8, 9, 10, 11, 12, 13, 14, 15);
            v16bf bb0 = *(const v16bf*)(wp + (size_t)c * NT * 512);
            v16bf bb1 = *(const v16bf*)(wp + (size_t)c * NT * 512 + 512);
            acc00 = __builtin_amdgcn_wmma_f32_16x16x32_bf16(
                false, a0, false, bb0, (short)0, acc00, false, false);
            acc01 = __builtin_amdgcn_wmma_f32_16x16x32_bf16(
                false, a0, false, bb1, (short)0, acc01, false, false);
            acc10 = __builtin_amdgcn_wmma_f32_16x16x32_bf16(
                false, a1, false, bb0, (short)0, acc10, false, false);
            acc11 = __builtin_amdgcn_wmma_f32_16x16x32_bf16(
                false, a1, false, bb1, (short)0, acc11, false, false);
        }
    }

    // D layout: lanes 0-15 -> M = r, lanes 16-31 -> M = 8+r; N = lane&15
    int nA = ntp * 32 + (lane & 15);
    int nB = nA + 16;
    float bsA = bias[nA];
    float bsB = bias[nB];
    int mbase0 = mtp * 32 + khalf * 8;
    int mbase1 = mbase0 + 16;
#pragma unroll
    for (int r = 0; r < 8; ++r) {
        float v00 = acc00[r] + bsA, v01 = acc01[r] + bsB;
        float v10 = acc10[r] + bsA, v11 = acc11[r] + bsB;
        if (relu) {
            v00 = v00 > 0.f ? v00 : 0.f;  v01 = v01 > 0.f ? v01 : 0.f;
            v10 = v10 > 0.f ? v10 : 0.f;  v11 = v11 > 0.f ? v11 : 0.f;
        }
        out[(size_t)(mbase0 + r) * out_cstride + nA] = (__bf16)v00;
        out[(size_t)(mbase0 + r) * out_cstride + nB] = (__bf16)v01;
        out[(size_t)(mbase1 + r) * out_cstride + nA] = (__bf16)v10;
        out[(size_t)(mbase1 + r) * out_cstride + nB] = (__bf16)v11;
    }
}

// ---------------------------------------------------------------------------
// Mask head: 1x1 conv 256->157 (+head bias + smpl_init), fp32 NHWC into d_out.
// GEMM M = B*g*g, N = 160 (157 padded), K = 256. One wave per 16x16 tile.
// ---------------------------------------------------------------------------
__global__ void __launch_bounds__(256)
head_gemm_wmma(const __bf16* __restrict__ in, int g, const __bf16* __restrict__ wf,
               const float* __restrict__ hb, const float* __restrict__ hinit,
               float* __restrict__ out) {
    const int NT = 10;
    int wid  = (blockIdx.x * blockDim.x + threadIdx.x) >> 5;
    int lane = threadIdx.x & 31;
    int M = BATCH * g * g;
    int W = (M / 16) * NT;
    if (wid >= W) return;   // wave-uniform: EXEC all-1s inside
    int mt = wid / NT, nt = wid % NT;
    int row = lane & 15, khalf = lane >> 4;
    const __bf16* rp = in + (size_t)(mt * 16 + row) * 256 + khalf * 16;

    v8f acc = {};
#pragma unroll
    for (int c = 0; c < 8; ++c) {
        v8bf lo = *(const v8bf*)(rp + c * 32);
        v8bf hi = *(const v8bf*)(rp + c * 32 + 8);
        v16bf a = __builtin_shufflevector(lo, hi,
            0, 1, 2, 3, 4, 5, 6, 7, 8, 9, 10, 11, 12, 13, 14, 15);
        v16bf bb = *(const v16bf*)(wf + (((size_t)c * NT + nt) * 32 + lane) * 16);
        acc = __builtin_amdgcn_wmma_f32_16x16x32_bf16(
            false, a, false, bb, (short)0, acc, false, false);
    }
    int n = nt * 16 + (lane & 15);
    if (n < 157) {
        float bs = hb[n] + hinit[n];
#pragma unroll
        for (int r = 0; r < 8; ++r) {
            int m = mt * 16 + khalf * 8 + r;
            out[(size_t)m * 157 + n] = acc[r] + bs;
        }
    }
}

// ---------------------------------------------------------------------------
// Category head: 3x3 conv 256->1 + bias + sigmoid. One thread per pixel.
// ---------------------------------------------------------------------------
__global__ void cate_head3x3(const __bf16* __restrict__ in,
                             const float* __restrict__ wgt,
                             const float* __restrict__ bias, int g,
                             float* __restrict__ heat) {
    int idx = blockIdx.x * blockDim.x + threadIdx.x;
    int total = BATCH * g * g;
    if (idx >= total) return;
    int x = idx % g;
    int y = (idx / g) % g;
    int b = idx / (g * g);
    float acc = bias[0];
    for (int tap = 0; tap < 9; ++tap) {
        int ny = y + tap / 3 - 1;
        int nx = x + tap % 3 - 1;
        if ((unsigned)ny < (unsigned)g && (unsigned)nx < (unsigned)g) {
            const __bf16* rp = in + (size_t)((b * g + ny) * g + nx) * 256;
            for (int ci = 0; ci < 256; ++ci)
                acc += (float)rp[ci] * wgt[ci * 9 + tap];
        }
    }
    heat[idx] = 1.0f / (1.0f + expf(-acc));
}

// points_nms: hmax over {y-1..y} x {x-1..x}; keep value only where it equals hmax
__global__ void nms_out(const float* __restrict__ heat, int g, float* __restrict__ out) {
    int idx = blockIdx.x * blockDim.x + threadIdx.x;
    int total = BATCH * g * g;
    if (idx >= total) return;
    int x = idx % g;
    int y = (idx / g) % g;
    float h = heat[idx];
    float m = h;
    if (y > 0)          m = fmaxf(m, heat[idx - g]);
    if (x > 0)          m = fmaxf(m, heat[idx - 1]);
    if (y > 0 && x > 0) m = fmaxf(m, heat[idx - g - 1]);
    out[idx] = (m == h) ? h : 0.0f;
}

// ---------------------------------------------------------------------------
extern "C" void kernel_launch(void* const* d_in, const int* in_sizes, int n_in,
                              void* d_out, int out_size, void* d_ws, size_t ws_size,
                              hipStream_t stream) {
    (void)in_sizes; (void)n_in; (void)out_size; (void)ws_size;

    const float* feat0       = (const float*)d_in[0];
    const float* feat1       = (const float*)d_in[1];
    const float* feat2       = (const float*)d_in[2];
    const float* feat3       = (const float*)d_in[3];
    const float* feat4       = (const float*)d_in[4];
    const float* smpl_w0     = (const float*)d_in[5];
    const float* smpl_b0     = (const float*)d_in[6];
    const float* smpl_w      = (const float*)d_in[7];   // (3,256,256,3,3)
    const float* smpl_b      = (const float*)d_in[8];   // (3,256)
    const float* cate_w0     = (const float*)d_in[9];
    const float* cate_b0     = (const float*)d_in[10];
    const float* cate_w      = (const float*)d_in[11];
    const float* cate_b      = (const float*)d_in[12];
    const float* cate_head_w = (const float*)d_in[13];
    const float* cate_head_b = (const float*)d_in[14];
    const float* smpl_head_w = (const float*)d_in[15];
    const float* smpl_head_b = (const float*)d_in[16];
    const float* smpl_init   = (const float*)d_in[17];
    float* out = (float*)d_out;

    // ---- workspace carve (all 256B aligned) ----
    uint8_t* wsp = (uint8_t*)d_ws;
    size_t off = 0;
    auto carve = [&](size_t bytes) -> void* {
        void* p = wsp + off;
        off += (bytes + 255) & ~(size_t)255;
        return p;
    };
    float*  f0     = (float*)carve((size_t)4 * 256 * 100 * 100 * 4);
    float*  f4     = (float*)carve((size_t)4 * 256 * 25 * 25 * 4);
    __bf16* actIn  = (__bf16*)carve((size_t)4 * 40 * 40 * 288 * 2);
    __bf16* actA   = (__bf16*)carve((size_t)4 * 40 * 40 * 256 * 2);
    __bf16* actB   = (__bf16*)carve((size_t)4 * 40 * 40 * 256 * 2);
    float*  heat   = (float*)carve((size_t)4 * 40 * 40 * 4);
    __bf16* zpage  = (__bf16*)carve((size_t)1024 * 2);            // zero page
    __bf16* pw_s0  = (__bf16*)carve((size_t)81 * 16 * 512 * 2);   // K=9*288
    __bf16* pw_s[3], *pw_c[3];
    for (int i = 0; i < 3; ++i) pw_s[i] = (__bf16*)carve((size_t)72 * 16 * 512 * 2);
    __bf16* pw_c0  = (__bf16*)carve((size_t)72 * 16 * 512 * 2);
    for (int i = 0; i < 3; ++i) pw_c[i] = (__bf16*)carve((size_t)72 * 16 * 512 * 2);
    __bf16* pw_h   = (__bf16*)carve((size_t)8 * 10 * 512 * 2);

    auto blocks = [](int total, int thr) { return (total + thr - 1) / thr; };

    // ---- pack all weights + zero page (deterministic, cheap) ----
    {
        zero_fill_bf16<<<blocks(1024, 256), 256, 0, stream>>>(zpage, 1024);
        int t0 = 81 * 16 * 512;
        pack_w3x3<<<blocks(t0, 256), 256, 0, stream>>>(smpl_w0, 258, 288, 81, 16, pw_s0);
        int t1 = 72 * 16 * 512;
        for (int i = 0; i < 3; ++i)
            pack_w3x3<<<blocks(t1, 256), 256, 0, stream>>>(
                smpl_w + (size_t)i * 256 * 256 * 9, 256, 256, 72, 16, pw_s[i]);
        pack_w3x3<<<blocks(t1, 256), 256, 0, stream>>>(cate_w0, 256, 256, 72, 16, pw_c0);
        for (int i = 0; i < 3; ++i)
            pack_w3x3<<<blocks(t1, 256), 256, 0, stream>>>(
                cate_w + (size_t)i * 256 * 256 * 9, 256, 256, 72, 16, pw_c[i]);
        int th = 8 * 10 * 512;
        pack_head<<<blocks(th, 256), 256, 0, stream>>>(smpl_head_w, pw_h);
    }

    // ---- pyramid pre-resizes ----
    resize_ac_nchw<<<blocks(1024 * 100 * 100, 256), 256, 0, stream>>>(
        feat0, 200, 200, 100, 100, 1024, f0);
    resize_ac_nchw<<<blocks(1024 * 25 * 25, 256), 256, 0, stream>>>(
        feat4, 13, 13, 25, 25, 1024, f4);

    const float* lsrc[5] = {f0, feat1, feat2, feat3, f4};
    const int    lh[5]   = {100, 100, 50, 25, 25};
    const int    lw[5]   = {100, 100, 50, 25, 25};
    const int    G[5]    = {40, 36, 24, 16, 12};

    size_t smpl_off[5], cate_off[5];
    {
        size_t o = 0;
        for (int l = 0; l < 5; ++l) { smpl_off[l] = o; o += (size_t)4 * G[l] * G[l] * 157; }
        for (int l = 0; l < 5; ++l) { cate_off[l] = o; o += (size_t)4 * G[l] * G[l]; }
    }

    for (int l = 0; l < 5; ++l) {
        int g = G[l];
        int M = 4 * g * g;          // multiple of 32 for every level
        int Mt = M / 16;

        resize_grid<<<blocks(M * 288, 256), 256, 0, stream>>>(lsrc[l], lh[l], lw[l], g, actIn);

        // conv grid: (Mt/2)*8 waves, 8 waves/block -> Mt/2 blocks
        int cblk = Mt / 2;

        // ---- mask (smpl) branch ----
        conv3x3_wmma<9><<<cblk, 256, 0, stream>>>(actIn, g, 288, pw_s0, zpage, smpl_b0, actA, 256, 1);
        conv3x3_wmma<8><<<cblk, 256, 0, stream>>>(actA, g, 256, pw_s[0], zpage, smpl_b + 0,   actB, 256, 1);
        conv3x3_wmma<8><<<cblk, 256, 0, stream>>>(actB, g, 256, pw_s[1], zpage, smpl_b + 256, actA, 256, 1);
        conv3x3_wmma<8><<<cblk, 256, 0, stream>>>(actA, g, 256, pw_s[2], zpage, smpl_b + 512, actB, 256, 1);
        {
            int hw = Mt * 10;                       // waves
            head_gemm_wmma<<<blocks(hw * 32, 256), 256, 0, stream>>>(
                actB, g, pw_h, smpl_head_b, smpl_init, out + smpl_off[l]);
        }

        // ---- category branch (reads first 256 channels of actIn) ----
        conv3x3_wmma<8><<<cblk, 256, 0, stream>>>(actIn, g, 288, pw_c0, zpage, cate_b0, actA, 256, 1);
        conv3x3_wmma<8><<<cblk, 256, 0, stream>>>(actA, g, 256, pw_c[0], zpage, cate_b + 0,   actB, 256, 1);
        conv3x3_wmma<8><<<cblk, 256, 0, stream>>>(actB, g, 256, pw_c[1], zpage, cate_b + 256, actA, 256, 1);
        conv3x3_wmma<8><<<cblk, 256, 0, stream>>>(actA, g, 256, pw_c[2], zpage, cate_b + 512, actB, 256, 1);
        cate_head3x3<<<blocks(M, 256), 256, 0, stream>>>(actB, cate_head_w, cate_head_b, g, heat);
        nms_out<<<blocks(M, 256), 256, 0, stream>>>(heat, g, out + cate_off[l]);
    }
}
